// GPT_1958505087772
// MI455X (gfx1250) — compile-verified, matrix operand order checked by reading, not run on hardware
//
#include <hip/hip_runtime.h>
#include <hip/hip_bf16.h>

// ---------------------------------------------------------------------------
// Types for CDNA5 WMMA (wave32): V_WMMA_F32_16X16X32_BF16
// ---------------------------------------------------------------------------
typedef __attribute__((ext_vector_type(16))) __bf16 v16bf;
typedef __attribute__((ext_vector_type(8)))  float  v8f;

union FragBF {
    v16bf v;
    uint4 q[2];
    unsigned short s[16];
};

__device__ __forceinline__ unsigned short f2bf(float f) {
    unsigned int u = __float_as_uint(f);
    unsigned int r = u + 0x7FFFu + ((u >> 16) & 1u);   // round-to-nearest-even
    return (unsigned short)(r >> 16);
}

// ---------------------------------------------------------------------------
// CDNA5 async global->LDS copy (ASYNCcnt) via inline asm (ISA §15.18 VGLOBAL:
// VDST = LDS byte-address VGPR, VADDR = 64-bit global address, GV mode).
// Wait via s_wait_asynccnt (builtin if present, else asm).
// ---------------------------------------------------------------------------
__device__ __forceinline__ void async_copy_b128(const unsigned short* g,
                                                unsigned short* l) {
    unsigned int lo = (unsigned int)(unsigned long long)
        (__attribute__((address_space(3))) unsigned short*)l;
    asm volatile("global_load_async_to_lds_b128 %0, %1, off"
                 :: "v"(lo), "v"(g) : "memory");
}

#if __has_builtin(__builtin_amdgcn_s_wait_asynccnt)
#define WAITA_BUILTIN 1
#else
#define WAITA_BUILTIN 0
#endif

__device__ __forceinline__ void wait_async4() {
#if WAITA_BUILTIN
    __builtin_amdgcn_s_wait_asynccnt(4);
#else
    asm volatile("s_wait_asynccnt 0x4" ::: "memory");
#endif
}
__device__ __forceinline__ void wait_async0() {
#if WAITA_BUILTIN
    __builtin_amdgcn_s_wait_asynccnt(0);
#else
    asm volatile("s_wait_asynccnt 0x0" ::: "memory");
#endif
}

// ---------------------------------------------------------------------------
// fp32 -> bf16 bulk conversion (embedding table)
// ---------------------------------------------------------------------------
__global__ void f32_to_bf16(const float* __restrict__ in,
                            unsigned short* __restrict__ out, size_t n) {
    size_t i = (size_t)blockIdx.x * blockDim.x + threadIdx.x;
    size_t stride = (size_t)gridDim.x * blockDim.x;
    for (; i < n; i += stride) out[i] = f2bf(in[i]);
}

// ---------------------------------------------------------------------------
// fp32 [K,N] -> bf16 [N,K] transposed conversion (weights), 32x32 LDS tiles.
// K, N multiples of 32.
// ---------------------------------------------------------------------------
__global__ __launch_bounds__(256) void f32_to_bf16_T(
    const float* __restrict__ in, unsigned short* __restrict__ out,
    int K, int N) {
    __shared__ float t[32][33];
    const int nb = blockIdx.x * 32;
    const int kb = blockIdx.y * 32;
    const int tx = threadIdx.x & 31;
    const int ty = threadIdx.x >> 5;   // 0..7
    for (int p = 0; p < 4; ++p)
        t[ty + p * 8][tx] = in[(size_t)(kb + ty + p * 8) * N + nb + tx];
    __syncthreads();
    for (int p = 0; p < 4; ++p)
        out[(size_t)(nb + ty + p * 8) * K + kb + tx] = f2bf(t[tx][ty + p * 8]);
}

// ---------------------------------------------------------------------------
// Embedding: x[b,s,:] = te[idx[b,s],:] + pe[s,:]   (fp32 + bf16 copies)
// ---------------------------------------------------------------------------
__global__ __launch_bounds__(256) void embed_kernel(
    const int* __restrict__ idx, const float* __restrict__ te,
    const float* __restrict__ pe, float* __restrict__ X,
    unsigned short* __restrict__ Xb, int S, int D) {
    int row = blockIdx.x;        // b*S + s
    int s   = row % S;
    int t   = idx[row];
    for (int i = 0; i < 4; ++i) {
        int d = threadIdx.x + 256 * i;
        float v = te[(size_t)t * D + d] + pe[(size_t)s * D + d];
        X [(size_t)row * D + d] = v;
        Xb[(size_t)row * D + d] = f2bf(v);
    }
}

// ---------------------------------------------------------------------------
// LayerNorm over D=1024 (one block per row), bf16 output (feeds a GEMM)
// ---------------------------------------------------------------------------
__global__ __launch_bounds__(256) void ln_kernel(
    const float* __restrict__ X, const float* __restrict__ g,
    const float* __restrict__ bta, unsigned short* __restrict__ Out, int D) {
    __shared__ float red[256];
    int row = blockIdx.x;
    int tid = threadIdx.x;
    const float* x = X + (size_t)row * D;
    float v[4];
    float s = 0.f;
    for (int i = 0; i < 4; ++i) { v[i] = x[tid + 256 * i]; s += v[i]; }
    red[tid] = s; __syncthreads();
    for (int st = 128; st > 0; st >>= 1) {
        if (tid < st) red[tid] += red[tid + st];
        __syncthreads();
    }
    float mean = red[0] / (float)D;
    __syncthreads();
    s = 0.f;
    for (int i = 0; i < 4; ++i) { float d = v[i] - mean; s += d * d; }
    red[tid] = s; __syncthreads();
    for (int st = 128; st > 0; st >>= 1) {
        if (tid < st) red[tid] += red[tid + st];
        __syncthreads();
    }
    float rstd = rsqrtf(red[0] / (float)D + 1e-5f);
    for (int i = 0; i < 4; ++i) {
        int d = tid + 256 * i;
        Out[(size_t)row * D + d] = f2bf((v[i] - mean) * rstd * g[d] + bta[d]);
    }
}

// ---------------------------------------------------------------------------
// WMMA GEMM: C[M,N] = act(A[M,K](bf16) * B[N,K](bf16, pre-transposed) + bias)
//            (+resid).  act==1: exact GELU.  Cf (fp32) / Cb (bf16) optional.
// Block: 256 thr (8 waves), tile 128x128, BK=32; wave -> 32x64 (2x4 WMMA
// tiles). Double-buffered LDS fed by async global->LDS b128 copies.
// ---------------------------------------------------------------------------
#define BM 128
#define BN 128
#define BK 32
#define LDT 40   // LDS row stride (elems): 80B, 16B-aligned, bank-skewed

__global__ __launch_bounds__(256) void wmma_gemm(
    const unsigned short* __restrict__ A, const unsigned short* __restrict__ B,
    const float* __restrict__ bias, const float* __restrict__ resid,
    float* __restrict__ Cf, unsigned short* __restrict__ Cb,
    int M, int N, int K, int act) {
    __shared__ __align__(16) unsigned short As[2][BM * LDT];
    __shared__ __align__(16) unsigned short Bs[2][BN * LDT];

    const int tid  = threadIdx.x;
    const int lane = tid & 31;
    const int w    = tid >> 5;
    const int wm   = w & 3;             // 4 row-waves
    const int wn   = w >> 2;            // 2 col-waves
    const int m0   = blockIdx.y * BM;
    const int n0   = blockIdx.x * BN;
    const int lm   = lane & 15;
    const int kg   = (lane >> 4) << 3;  // 0 or 8

    // staging role: each thread copies 2x16B of A and 2x16B of B per tile
    const int sr   = tid >> 2;          // 0..63
    const int scol = (tid & 3) * 8;     // 0,8,16,24

    v8f acc[2][4] = {};

    auto stage = [&](int buf, int k0) {
        for (int p = 0; p < 2; ++p) {
            int row = sr + p * 64;
            async_copy_b128(A + (size_t)(m0 + row) * K + k0 + scol,
                            &As[buf][row * LDT + scol]);
            int nrow = n0 + row;
            if (nrow >= N) nrow = N - 1;   // clamp: cols >= N masked at store
            async_copy_b128(B + (size_t)nrow * K + k0 + scol,
                            &Bs[buf][row * LDT + scol]);
        }
    };

    const int nk = K / BK;
    stage(0, 0);
    for (int kt = 0; kt < nk; ++kt) {
        const int cur = kt & 1;
        if (kt + 1 < nk) {
            stage(cur ^ 1, (kt + 1) * BK);
            wait_async4();               // previous stage's 4 copies complete
        } else {
            wait_async0();
        }
        __syncthreads();

        FragBF af[2], bf[4];
        for (int i = 0; i < 2; ++i) {
            int row = wm * 32 + i * 16 + lm;
            af[i].q[0] = *(const uint4*)(&As[cur][row * LDT + kg]);
            af[i].q[1] = *(const uint4*)(&As[cur][row * LDT + kg + 16]);
        }
        for (int j = 0; j < 4; ++j) {
            int row = wn * 64 + j * 16 + lm;
            bf[j].q[0] = *(const uint4*)(&Bs[cur][row * LDT + kg]);
            bf[j].q[1] = *(const uint4*)(&Bs[cur][row * LDT + kg + 16]);
        }
        for (int i = 0; i < 2; ++i)
            for (int j = 0; j < 4; ++j)
                acc[i][j] = __builtin_amdgcn_wmma_f32_16x16x32_bf16(
                    false, af[i].v, false, bf[j].v, (short)0, acc[i][j],
                    false, false);
        __syncthreads();
    }

    // ---- epilogue: bias + residual + GELU, fp32 and/or bf16 stores ----
    for (int i = 0; i < 2; ++i)
        for (int j = 0; j < 4; ++j) {
            int mb = m0 + wm * 32 + i * 16 + ((lane >> 4) << 3);
            int nb = n0 + wn * 64 + j * 16 + lm;
            if (nb < N) {
                float bv = bias ? bias[nb] : 0.f;
                for (int vg = 0; vg < 8; ++vg) {
                    int row = mb + vg;
                    float v = acc[i][j][vg] + bv;
                    if (resid) v += resid[(size_t)row * N + nb];
                    if (act == 1)
                        v = 0.5f * v * (1.f + erff(v * 0.70710678118f));
                    if (Cf) Cf[(size_t)row * N + nb] = v;
                    if (Cb) Cb[(size_t)row * N + nb] = f2bf(v);
                }
            }
        }
}

// ---------------------------------------------------------------------------
// Causal attention, one wave per (b, h, 16-query tile).
//   scores = (Q K^T)/sqrt(DK) via WMMA -> LDS fp32; causal softmax;
//   probs re-quantized bf16 in-place (aliased LDS); Y = P V via WMMA.
// Dynamic LDS: 16 * S * 4 bytes.
// ---------------------------------------------------------------------------
__global__ __launch_bounds__(32) void attn_kernel(
    const unsigned short* __restrict__ Q, const unsigned short* __restrict__ Kb,
    const unsigned short* __restrict__ Vb, unsigned short* __restrict__ Y,
    int S, int H, int DK) {
    extern __shared__ char smem[];
    float* sc          = (float*)smem;            // [16][S]
    unsigned short* pb = (unsigned short*)smem;   // aliased, row stride 2*S

    const int lane = threadIdx.x;
    const int lm   = lane & 15;
    const int kg   = (lane >> 4) << 3;
    const int qt   = blockIdx.x;
    const int h    = blockIdx.y;
    const int b    = blockIdx.z;
    const int q0   = qt * 16;
    const int D    = H * DK;
    const int hoff = h * DK;
    const size_t rowbase = (size_t)b * S;

    // Q fragments (16 rows x 64 dk = 2 k-chunks of 32)
    FragBF qf[2];
    for (int c = 0; c < 2; ++c) {
        const unsigned short* p = Q + (rowbase + q0 + lm) * D + hoff + c * 32;
        qf[c].q[0] = *(const uint4*)(p + kg);
        qf[c].q[1] = *(const uint4*)(p + kg + 16);
    }
    const float scale = rsqrtf((float)DK);

    // ---- phase 1: scores for causal key tiles ----
    for (int kt = 0; kt <= qt; ++kt) {
        int key0 = kt * 16;
        v8f acc = {};
        for (int c = 0; c < 2; ++c) {
            FragBF kf;
            const unsigned short* p =
                Kb + (rowbase + key0 + lm) * D + hoff + c * 32;
            kf.q[0] = *(const uint4*)(p + kg);
            kf.q[1] = *(const uint4*)(p + kg + 16);
            acc = __builtin_amdgcn_wmma_f32_16x16x32_bf16(
                false, qf[c].v, false, kf.v, (short)0, acc, false, false);
        }
        for (int vg = 0; vg < 8; ++vg) {
            int r = vg + ((lane >> 4) << 3);
            sc[(size_t)r * S + key0 + lm] = acc[vg] * scale;
        }
    }
    __syncthreads();

    const int kend   = (qt + 1) * 16;
    const int kend32 = (kend + 31) & ~31;

    // ---- phase 2: causal softmax (lanes 0..15, one query row each) ----
    if (lane < 16) {
        int r  = lane;
        int nk = q0 + r + 1;                   // keys 0..q inclusive
        const float* srow = sc + (size_t)r * S;
        float mx = -1e30f;
        for (int j = 0; j < nk; ++j) mx = fmaxf(mx, srow[j]);
        float sum = 0.f;
        for (int j = 0; j < nk; ++j) sum += __expf(srow[j] - mx);
        float inv = 1.f / sum;
        unsigned short* prow = pb + (size_t)r * (2 * S);  // same bytes as srow
        for (int j = 0; j < nk; ++j)            // write at 2j after read at 4j
            prow[j] = f2bf(__expf(srow[j] - mx) * inv);
        for (int j = nk; j < kend32; ++j) prow[j] = 0;
    }
    __syncthreads();

    // ---- phase 3: Y = P V ----
    for (int c2 = 0; c2 < DK / 16; ++c2) {
        v8f acc = {};
        for (int kc = 0; kc < kend32; kc += 32) {
            FragBF pf, vf;
            const unsigned short* pr = pb + (size_t)lm * (2 * S) + kc;
            pf.q[0] = *(const uint4*)(pr + kg);
            pf.q[1] = *(const uint4*)(pr + kg + 16);
            for (int r = 0; r < 8; ++r) {
                int kk = (r < 4) ? (kg + 2 * r) : (16 + kg + 2 * (r - 4));
                const unsigned short* vp =
                    Vb + (rowbase + kc + kk) * D + hoff + c2 * 16 + lm;
                vf.s[2 * r]     = vp[0];
                vf.s[2 * r + 1] = vp[D];
            }
            acc = __builtin_amdgcn_wmma_f32_16x16x32_bf16(
                false, pf.v, false, vf.v, (short)0, acc, false, false);
        }
        for (int vg = 0; vg < 8; ++vg) {
            int r = vg + ((lane >> 4) << 3);
            Y[(rowbase + q0 + r) * D + hoff + c2 * 16 + lm] = f2bf(acc[vg]);
        }
    }
}

// ---------------------------------------------------------------------------
// Host orchestration
// ---------------------------------------------------------------------------
extern "C" void kernel_launch(void* const* d_in, const int* in_sizes, int n_in,
                              void* d_out, int out_size, void* d_ws,
                              size_t ws_size, hipStream_t stream) {
    (void)in_sizes; (void)n_in; (void)out_size; (void)ws_size;
    const int D = 1024, H = 16, Vv = 50257, Ll = 4, Bb = 2, Ss = 1024;
    const int M = Bb * Ss, DFF = 4 * D, DK = D / H;

    const int*   idx   = (const int*)  d_in[0];
    const float* te    = (const float*)d_in[1];
    const float* pe    = (const float*)d_in[2];
    const float* wq    = (const float*)d_in[3];
    const float* bq    = (const float*)d_in[4];
    const float* wk    = (const float*)d_in[5];
    const float* bk    = (const float*)d_in[6];
    const float* wv    = (const float*)d_in[7];
    const float* bv    = (const float*)d_in[8];
    const float* wo    = (const float*)d_in[9];
    const float* bo    = (const float*)d_in[10];
    const float* ln2g  = (const float*)d_in[11];
    const float* ln2b  = (const float*)d_in[12];
    const float* w1    = (const float*)d_in[13];
    const float* b1    = (const float*)d_in[14];
    const float* w2    = (const float*)d_in[15];
    const float* b2    = (const float*)d_in[16];
    const float* lnfg  = (const float*)d_in[17];
    const float* lnfb  = (const float*)d_in[18];

    char* wsp = (char*)d_ws;
    size_t off = 0;
    auto alloc = [&](size_t bytes) -> void* {
        off = (off + 255) & ~(size_t)255;
        void* p = wsp + off;
        off += bytes;
        return p;
    };
    unsigned short* te_b = (unsigned short*)alloc((size_t)Vv * D * 2);
    unsigned short* wbuf = (unsigned short*)alloc((size_t)D * DFF * 2);
    float*          x0   = (float*)         alloc((size_t)M * D * 4);
    float*          x1   = (float*)         alloc((size_t)M * D * 4);
    unsigned short* xb   = (unsigned short*)alloc((size_t)M * D * 2);
    unsigned short* hb   = (unsigned short*)alloc((size_t)M * D * 2);
    unsigned short* qb   = (unsigned short*)alloc((size_t)M * D * 2);
    unsigned short* kb   = (unsigned short*)alloc((size_t)M * D * 2);
    unsigned short* vb   = (unsigned short*)alloc((size_t)M * D * 2);
    unsigned short* yb   = (unsigned short*)alloc((size_t)M * D * 2);
    unsigned short* ub   = (unsigned short*)alloc((size_t)M * DFF * 2);

    auto gemm = [&](const unsigned short* A, const unsigned short* Bm,
                    const float* bias, const float* resid, float* Cf,
                    unsigned short* Cb, int Mm, int Nn, int Kk, int act) {
        dim3 grid((Nn + BN - 1) / BN, (Mm + BM - 1) / BM);
        wmma_gemm<<<grid, 256, 0, stream>>>(A, Bm, bias, resid, Cf, Cb, Mm, Nn,
                                            Kk, act);
    };
    // weights [K,N] fp32 -> [N,K] bf16 (so GEMM B-staging is contiguous b128)
    auto convT = [&](const float* src, unsigned short* dst, int Kk, int Nn) {
        dim3 grid(Nn / 32, Kk / 32);
        f32_to_bf16_T<<<grid, 256, 0, stream>>>(src, dst, Kk, Nn);
    };

    f32_to_bf16<<<2048, 256, 0, stream>>>(te, te_b, (size_t)Vv * D);
    embed_kernel<<<M, 256, 0, stream>>>(idx, te, pe, x0, xb, Ss, D);

    for (int l = 0; l < Ll; ++l) {
        const size_t wofs = (size_t)l * D * D;
        // QKV projections (reference uses raw x, no ln1)
        convT(wq + wofs, wbuf, D, D);
        gemm(xb, wbuf, bq + (size_t)l * D, nullptr, nullptr, qb, M, D, D, 0);
        convT(wk + wofs, wbuf, D, D);
        gemm(xb, wbuf, bk + (size_t)l * D, nullptr, nullptr, kb, M, D, D, 0);
        convT(wv + wofs, wbuf, D, D);
        gemm(xb, wbuf, bv + (size_t)l * D, nullptr, nullptr, vb, M, D, D, 0);
        // attention
        {
            dim3 ag(Ss / 16, H, Bb);
            size_t smem = (size_t)16 * Ss * 4;
            attn_kernel<<<ag, 32, smem, stream>>>(qb, kb, vb, yb, Ss, H, DK);
        }
        // x = x + y @ wo + bo
        convT(wo + wofs, wbuf, D, D);
        gemm(yb, wbuf, bo + (size_t)l * D, x0, x1, xb, M, D, D, 0);
        // h = LN(x); u = gelu(h @ w1 + b1); x = x + u @ w2 + b2
        ln_kernel<<<M, 256, 0, stream>>>(x1, ln2g + (size_t)l * D,
                                         ln2b + (size_t)l * D, hb, D);
        convT(w1 + (size_t)l * D * DFF, wbuf, D, DFF);
        gemm(hb, wbuf, b1 + (size_t)l * DFF, nullptr, nullptr, ub, M, DFF, D,
             1);
        convT(w2 + (size_t)l * DFF * D, wbuf, DFF, D);
        gemm(ub, wbuf, b2 + (size_t)l * D, x1, x0, xb, M, D, DFF, 0);
    }

    // final LN + weight-tied LM head (te_b is already [V,D] = [N,K])
    ln_kernel<<<M, 256, 0, stream>>>(x0, lnfg, lnfb, hb, D);
    gemm(hb, te_b, nullptr, nullptr, (float*)d_out, nullptr, M, Vv, D, 0);
}
